// ProbabilisticLayer_54666343743696
// MI455X (gfx1250) — matrix operand order repaired and checked by reading, not compiled, not run on hardware
//
#include <hip/hip_runtime.h>
#include <hip/hip_bf16.h>
#include <stdint.h>

// ---------------- problem constants ----------------
#define B_ 2
#define S_ 2048
#define H_ 1024
#define M_ (B_ * S_)              // 4096 rows for the QKV projection
#define BSH ((size_t)B_ * S_ * H_)
#define NSAMP 64
#define DROPP 0.1f
#define KEEPP 0.9f
#define SCALE 0.3125f             // 1/(sqrt(1024)*0.1)

typedef __attribute__((ext_vector_type(16))) _Float16 v16h;
typedef __attribute__((ext_vector_type(8)))  _Float16 v8h;
typedef __attribute__((ext_vector_type(8)))  float    v8f;

// ---------------- counter-based RNG (reproducible across passes) ----------------
__device__ __forceinline__ uint32_t hash_u64(uint64_t x) {
  x ^= x >> 33; x *= 0xff51afd7ed558ccdULL;
  x ^= x >> 33; x *= 0xc4ceb9fe1a85ec53ULL;
  x ^= x >> 33;
  return (uint32_t)x;
}

__device__ __forceinline__ bool keep_elem(unsigned kind, unsigned sample, uint64_t idx) {
  uint64_t key = ((uint64_t)(sample * 4u + kind + 1u) << 36) ^ idx ^ 0x9E3779B97F4A7C15ULL;
  return (hash_u64(key) >> 8) < 15099494u; // 0.9 * 2^24
}

__device__ __forceinline__ float gumbelf(unsigned sample, int b, int q, int k) {
  uint64_t key = 0x0123456789ABCDEFULL
               ^ ((uint64_t)(sample * 2u + (unsigned)b + 1u) << 44)
               ^ ((uint64_t)(unsigned)q << 22)
               ^ (uint64_t)(unsigned)k;
  float u = (float)(hash_u64(key) >> 8) * (1.0f / 16777216.0f);
  return -__logf(-__logf(u + 1e-10f));
}

// ---------------- WMMA fragment loaders (16-bit, per CDNA5 ISA layouts) ----------------
// A 16x32: lanes 0-15 -> M=0..15 (K=0-7,16-23); lanes 16-31 -> M=0..15 (K=8-15,24-31)
__device__ __forceinline__ v16h load_a_frag(const _Float16* base, int ld, int row0, int k0, int lid) {
  int row = row0 + (lid & 15);
  int hi  = (lid >> 4) * 8;
  const _Float16* p = base + (size_t)row * ld + k0 + hi;
  union { v16h v; v8h h[2]; } u;
  u.h[0] = *(const v8h*)(p);
  u.h[1] = *(const v8h*)(p + 16);
  return u.v;
}

// B 32x16: lane n = lid&15; lanes 0-15 hold K=0-7,16-23; lanes 16-31 hold K=8-15,24-31.
// Source matrix stored row-major over K per n (i.e. M[n*ld + k]) -> two runs of 8 contiguous halves, 16 apart.
__device__ __forceinline__ v16h load_b_frag(const _Float16* base, int ld, int n0, int k0, int lid) {
  int n  = n0 + (lid & 15);
  int kk = k0 + (lid >> 4) * 8;
  const _Float16* p = base + (size_t)n * ld + kk;
  union { v16h v; v8h h[2]; } u;
  u.h[0] = *(const v8h*)(p);
  u.h[1] = *(const v8h*)(p + 16);
  return u.v;
}

// Four adjacent 16x16 score tiles sharing one A-fragment load per K step:
// scores[q, k0+t*16 ..] = sum_h Qs[q,h]*Ks[k,h]. Each tile keeps its own
// accumulator chain in identical h-order -> bit-identical across passes.
__device__ __forceinline__ void score_tile4(const _Float16* Qb, const _Float16* Kb,
                                            int q0, int k0, int lid, v8f acc[4]) {
  v8f z = {};
  acc[0] = z; acc[1] = z; acc[2] = z; acc[3] = z;
  for (int h0 = 0; h0 < H_; h0 += 32) {
    v16h a = load_a_frag(Qb, H_, q0, h0, lid);
#pragma unroll
    for (int t = 0; t < 4; ++t) {
      v16h b = load_b_frag(Kb, H_, k0 + t * 16, h0, lid);
      acc[t] = __builtin_amdgcn_wmma_f32_16x16x32_f16(false, a, false, b, (short)0, acc[t], false, false);
    }
  }
}

// ---------------- elementwise kernels ----------------
__global__ void k_zero(float* p, long long n) {
  long long i = (long long)blockIdx.x * blockDim.x + threadIdx.x;
  long long st = (long long)gridDim.x * blockDim.x;
  for (; i < n; i += st) p[i] = 0.0f;
}

__global__ void k_cvt(const float* in, _Float16* out, long long n) {
  long long i = (long long)blockIdx.x * blockDim.x + threadIdx.x;
  long long st = (long long)gridDim.x * blockDim.x;
  for (; i < n; i += st) out[i] = (_Float16)in[i];
}

__global__ void k_drop(const _Float16* in, _Float16* out, long long n,
                       unsigned kind, unsigned sample) {
  long long i = (long long)blockIdx.x * blockDim.x + threadIdx.x;
  long long st = (long long)gridDim.x * blockDim.x;
  const float inv = 1.0f / KEEPP;
  for (; i < n; i += st) {
    bool k = keep_elem(kind, sample, (uint64_t)i);
    out[i] = k ? (_Float16)((float)in[i] * inv) : (_Float16)0.0f;
  }
}

// V [B,S,H] -> dropped + transposed Vt [B,H,S] so PV B-fragments are contiguous.
__global__ void k_drop_vt(const _Float16* V, _Float16* Vt, unsigned sample) {
  long long n = (long long)B_ * H_ * S_;
  long long i = (long long)blockIdx.x * blockDim.x + threadIdx.x;
  long long st = (long long)gridDim.x * blockDim.x;
  const float inv = 1.0f / KEEPP;
  for (; i < n; i += st) {
    int b = (int)(i / ((long long)H_ * S_));
    long long rem = i % ((long long)H_ * S_);
    int h = (int)(rem / S_);
    int k = (int)(rem % S_);
    long long src = (long long)b * S_ * H_ + (long long)k * H_ + h;
    bool kp = keep_elem(2u, sample, (uint64_t)src);
    Vt[i] = kp ? (_Float16)((float)V[src] * inv) : (_Float16)0.0f;
  }
}

// ---------------- QKV projection: out[m,o] = sum_h X[m,h]*W[o,h] + bias[o] ----------------
__global__ __launch_bounds__(32) void k_proj(const _Float16* X, const _Float16* W,
                                             const float* bias, _Float16* out) {
  int lid = threadIdx.x & 31;
  int m0 = blockIdx.x * 16;
  int n0 = blockIdx.y * 16;
  v8f acc = {};
  for (int k0 = 0; k0 < H_; k0 += 32) {
    v16h a = load_a_frag(X, H_, m0, k0, lid);
    v16h b = load_b_frag(W, H_, n0, k0, lid);
    acc = __builtin_amdgcn_wmma_f32_16x16x32_f16(false, a, false, b, (short)0, acc, false, false);
  }
  int col = n0 + (lid & 15);
  float bv = bias[col];
  int rbase = m0 + ((lid >> 4) << 3);
  for (int r = 0; r < 8; ++r)
    out[(size_t)(rbase + r) * H_ + col] = (_Float16)(acc[r] + bv);
}

// ---------------- Pass A: per-row softmax max & sum-of-exp (flash style) ----------------
// 8 waves per 16-row q-block; each wave owns k-tile groups g, g+8, ... (4 tiles/group).
__global__ __launch_bounds__(256) void k_stats(const _Float16* Qs, const _Float16* Ks,
                                               float* mout, float* lout, unsigned sample) {
  __shared__ float sm[8][16];
  __shared__ float sl[8][16];
  int b = blockIdx.y;
  int q0 = blockIdx.x * 16;
  int wave = threadIdx.x >> 5, lid = threadIdx.x & 31;
  int hi = lid >> 4;
  const _Float16* Qb = Qs + (size_t)b * S_ * H_;
  const _Float16* Kb = Ks + (size_t)b * S_ * H_;

  float mrun[8], lrun[8];
  for (int r = 0; r < 8; ++r) { mrun[r] = -3.0e38f; lrun[r] = 0.0f; }

  for (int g = wave; g < S_ / 64; g += 8) {
    v8f acc[4];
    score_tile4(Qb, Kb, q0, g * 64, lid, acc);
#pragma unroll
    for (int t = 0; t < 4; ++t) {
      int kcol = g * 64 + t * 16 + (lid & 15);
      for (int r = 0; r < 8; ++r) {
        int qrow = q0 + r + hi * 8;
        float sc = acc[t][r] * SCALE + gumbelf(sample, b, qrow, kcol);
        float mx = sc;
        for (int off = 1; off < 16; off <<= 1) mx = fmaxf(mx, __shfl_xor(mx, off, 32));
        float mnew = fmaxf(mrun[r], mx);
        float e = __expf(sc - mnew);
        for (int off = 1; off < 16; off <<= 1) e += __shfl_xor(e, off, 32);
        lrun[r] = lrun[r] * __expf(mrun[r] - mnew) + e;
        mrun[r] = mnew;
      }
    }
  }
  if ((lid & 15) == 0)
    for (int r = 0; r < 8; ++r) { sm[wave][r + hi * 8] = mrun[r]; sl[wave][r + hi * 8] = lrun[r]; }
  __syncthreads();
  if (threadIdx.x < 16) {
    int row = threadIdx.x;
    float M = -3.0e38f;
    for (int w = 0; w < 8; ++w) M = fmaxf(M, sm[w][row]);
    float L = 0.0f;
    for (int w = 0; w < 8; ++w) L += sl[w][row] * __expf(sm[w][row] - M);
    mout[(size_t)b * S_ + q0 + row] = M;
    lout[(size_t)b * S_ + q0 + row] = L;
  }
}

// ---------------- Pass B: out = softmax(scores)*Vs, accumulate sum & sum^2 ----------------
// 8 waves per q-block; k super-block of 512 per iteration. Wave w computes 4 score
// tiles (k0 = it*512 + w*64), writes P=exp(s-m) f16 into a shared 16x512 tile; then
// every wave runs PV WMMAs against its private 128-wide H slice of Vt, amortizing
// each P A-fragment over 8 h-tile WMMAs.
__global__ __launch_bounds__(256) void k_out(const _Float16* Qs, const _Float16* Ks,
                                             const _Float16* Vt,
                                             const float* mbuf, const float* lbuf,
                                             float* dsum, float* dss, unsigned sample) {
  __shared__ _Float16 pt[16 * 512];   // 16 KB
  int b = blockIdx.y;
  int q0 = blockIdx.x * 16;
  int wave = threadIdx.x >> 5, lid = threadIdx.x & 31;
  int hi = lid >> 4;
  const _Float16* Qb = Qs + (size_t)b * S_ * H_;
  const _Float16* Kb = Ks + (size_t)b * S_ * H_;
  const _Float16* Vb = Vt + (size_t)b * H_ * S_;   // [H][S]

  float mrow[8], linv[8];
  for (int r = 0; r < 8; ++r) {
    int qrow = q0 + r + hi * 8;
    mrow[r] = mbuf[(size_t)b * S_ + qrow];
    linv[r] = 1.0f / lbuf[(size_t)b * S_ + qrow];
  }

  v8f oacc[8] = {};
  int hbase = wave * 128;

  for (int it = 0; it < S_ / 512; ++it) {
    int k0 = it * 512 + wave * 64;
    v8f acc[4];
    score_tile4(Qb, Kb, q0, k0, lid, acc);   // identical per-tile chains to pass A
#pragma unroll
    for (int t = 0; t < 4; ++t) {
      int kcol = k0 + t * 16 + (lid & 15);
      for (int r = 0; r < 8; ++r) {
        int qrow = q0 + r + hi * 8;
        float sc = acc[t][r] * SCALE + gumbelf(sample, b, qrow, kcol);
        float p = __expf(sc - mrow[r]);
        pt[(r + hi * 8) * 512 + (wave * 64 + t * 16 + (lid & 15))] = (_Float16)p;
      }
    }
    __syncthreads();
    for (int kk = 0; kk < 16; ++kk) {
      v16h a = load_a_frag(pt, 512, 0, kk * 32, lid);
#pragma unroll
      for (int t = 0; t < 8; ++t) {
        v16h bf = load_b_frag(Vb, S_, hbase + t * 16, it * 512 + kk * 32, lid);
        oacc[t] = __builtin_amdgcn_wmma_f32_16x16x32_f16(false, a, false, bf, (short)0, oacc[t], false, false);
      }
    }
    __syncthreads();
  }

  for (int t = 0; t < 8; ++t) {
    int hcol = hbase + t * 16 + (lid & 15);
    for (int r = 0; r < 8; ++r) {
      int qrow = q0 + r + hi * 8;
      float v = oacc[t][r] * linv[r];
      size_t idx = ((size_t)b * S_ + qrow) * H_ + hcol;
      dsum[idx] += v;
      dss[idx]  += v * v;
    }
  }
}

// ---------------- finalize mean / std in place ----------------
__global__ void k_final(float* out, long long n) {
  long long i = (long long)blockIdx.x * blockDim.x + threadIdx.x;
  long long st = (long long)gridDim.x * blockDim.x;
  for (; i < n; i += st) {
    float s  = out[i];
    float ss = out[n + i];
    float mean = s * (1.0f / (float)NSAMP);
    float var = (ss - (float)NSAMP * mean * mean) * (1.0f / (float)(NSAMP - 1));
    out[i] = mean;
    out[n + i] = sqrtf(fmaxf(var, 0.0f));
  }
}

// ---------------- host-side launch ----------------
extern "C" void kernel_launch(void* const* d_in, const int* in_sizes, int n_in,
                              void* d_out, int out_size, void* d_ws, size_t ws_size,
                              hipStream_t stream) {
  const float* x  = (const float*)d_in[0];
  const float* wq = (const float*)d_in[1];
  const float* bq = (const float*)d_in[2];
  const float* wk = (const float*)d_in[3];
  const float* bk = (const float*)d_in[4];
  const float* wv = (const float*)d_in[5];
  const float* bv = (const float*)d_in[6];
  float* out = (float*)d_out;

  // workspace layout (f16): xh | wh(3) | Q K V | Qs Ks Vt | m l   (~65 MB)
  _Float16* xh = (_Float16*)d_ws;
  _Float16* wh = xh + (size_t)M_ * H_;
  _Float16* Qb = wh + 3ull * H_ * H_;
  _Float16* Kb = Qb + BSH;
  _Float16* Vb = Kb + BSH;
  _Float16* Qs = Vb + BSH;
  _Float16* Ks = Qs + BSH;
  _Float16* Vt = Ks + BSH;
  float* mbuf = (float*)(Vt + BSH);
  float* lbuf = mbuf + (size_t)B_ * S_;

  k_zero<<<2048, 256, 0, stream>>>(out, 2ll * (long long)BSH);
  k_cvt<<<2048, 256, 0, stream>>>(x, xh, (long long)M_ * H_);
  k_cvt<<<1024, 256, 0, stream>>>(wq, wh,                      (long long)H_ * H_);
  k_cvt<<<1024, 256, 0, stream>>>(wk, wh + (size_t)H_ * H_,    (long long)H_ * H_);
  k_cvt<<<1024, 256, 0, stream>>>(wv, wh + 2ull * H_ * H_,     (long long)H_ * H_);

  dim3 gp(M_ / 16, H_ / 16);
  k_proj<<<gp, 32, 0, stream>>>(xh, wh,                   bq, Qb);
  k_proj<<<gp, 32, 0, stream>>>(xh, wh + (size_t)H_ * H_, bk, Kb);
  k_proj<<<gp, 32, 0, stream>>>(xh, wh + 2ull * H_ * H_,  bv, Vb);

  dim3 ga(S_ / 16, B_);
  for (unsigned s = 0; s < NSAMP; ++s) {
    k_drop<<<2048, 256, 0, stream>>>(Qb, Qs, (long long)BSH, 0u, s);
    k_drop<<<2048, 256, 0, stream>>>(Kb, Ks, (long long)BSH, 1u, s);
    k_drop_vt<<<2048, 256, 0, stream>>>(Vb, Vt, s);
    k_stats<<<ga, 256, 0, stream>>>(Qs, Ks, mbuf, lbuf, s);
    k_out<<<ga, 256, 0, stream>>>(Qs, Ks, Vt, mbuf, lbuf, out, out + BSH, s);
  }
  k_final<<<2048, 256, 0, stream>>>(out, (long long)BSH);
}
